// Target_encoder_9826885173488
// MI455X (gfx1250) — compile-verified, matrix operand order checked by reading, not run on hardware
//
#include <hip/hip_runtime.h>
#include <hip/hip_bf16.h>
#include <cstdint>

typedef __attribute__((ext_vector_type(16))) _Float16 v16h;
typedef __attribute__((ext_vector_type(8)))  float    v8f;

#define BN 8
#define LL 1024
#define DD 128
#define NFEAT 33
#define HH 4
#define DK 16
#define NN (BN*LL)

// ---------------- WMMA helpers (layouts per CDNA5 ISA 7.12.2) ----------------

static __device__ __forceinline__ v8f wmma_f16(v16h a, v16h b, v8f c) {
  // D = A(16x32 f16) * B(32x16 f16) + C(16x16 f32)
  return __builtin_amdgcn_wmma_f32_16x16x32_f16(false, a, false, b, (short)0, c, false, false);
}

// A fragment: 16x32 f16, source row-major A[m][k] with leading dim ld.
// Lane t: m = t&15, hi = t>>4. Element e -> k = 16*(e>>3) + 8*hi + (e&7).
static __device__ __forceinline__ v16h load_a_frag(const _Float16* A, int ld, int lane,
                                                   int kbase, int kmax) {
  int m = lane & 15, hi = (lane >> 4) & 1;
  v16h a;
#pragma unroll
  for (int e = 0; e < 16; ++e) {
    int k = kbase + ((e >> 3) << 4) + (hi << 3) + (e & 7);
    a[e] = (k < kmax) ? A[m * ld + k] : (_Float16)0.0f;
  }
  return a;
}

// B fragment: 32x16 f16, source "BT" layout BT[n][k] (i.e. B transposed, K contiguous).
static __device__ __forceinline__ v16h load_b_frag(const _Float16* BT, int ld, int lane,
                                                   int kbase, int kmax, int nmax) {
  int n = lane & 15, hi = (lane >> 4) & 1;
  v16h b;
#pragma unroll
  for (int e = 0; e < 16; ++e) {
    int k = kbase + ((e >> 3) << 4) + (hi << 3) + (e & 7);
    b[e] = (n < nmax && k < kmax) ? BT[n * ld + k] : (_Float16)0.0f;
  }
  return b;
}

static __device__ __forceinline__ float red16_max(float v) {
#pragma unroll
  for (int off = 8; off >= 1; off >>= 1) v = fmaxf(v, __shfl_xor(v, off, 32));
  return v;
}
static __device__ __forceinline__ float red16_sum(float v) {
#pragma unroll
  for (int off = 8; off >= 1; off >>= 1) v += __shfl_xor(v, off, 32);
  return v;
}
static __device__ __forceinline__ float red32_sum(float v) {
#pragma unroll
  for (int off = 16; off >= 1; off >>= 1) v += __shfl_xor(v, off, 32);
  return v;
}

// ---------------- GCN pieces ----------------

__global__ void gcn1_xw_kernel(const float* __restrict__ px, const float* __restrict__ W,
                               float* __restrict__ xw, int total) {
  int idx = blockIdx.x * blockDim.x + threadIdx.x;
  if (idx >= total) return;
  int n = idx >> 7, d = idx & 127;
  float acc = 0.f;
  for (int f = 0; f < NFEAT; ++f) acc += px[n * NFEAT + f] * W[f * DD + d];
  xw[idx] = acc;
}

__global__ void deg_init_kernel(float* deg) {
  int n = blockIdx.x * blockDim.x + threadIdx.x;
  if (n < NN) deg[n] = 1.0f;  // self-loop
}
__global__ void deg_accum_kernel(const int* __restrict__ dst, float* deg, int E) {
  int e = blockIdx.x * blockDim.x + threadIdx.x;
  if (e < E) atomicAdd(&deg[dst[e]], 1.0f);
}

__global__ void gcn_selfbias_kernel(const float* __restrict__ xw, const float* __restrict__ deg,
                                    const float* __restrict__ bias, float* __restrict__ out,
                                    int total) {
  int idx = blockIdx.x * blockDim.x + threadIdx.x;
  if (idx >= total) return;
  int n = idx >> 7, d = idx & 127;
  out[idx] = xw[idx] / deg[n] + bias[d];  // self-loop weight = dinv^2 = 1/deg
}

__global__ void gcn_scatter_kernel(const int* __restrict__ src, const int* __restrict__ dst,
                                   const float* __restrict__ xw, const float* __restrict__ deg,
                                   float* __restrict__ out, int E) {
  int gid = blockIdx.x * blockDim.x + threadIdx.x;
  int e = gid >> 5, lane = gid & 31;
  if (e >= E) return;
  int s = src[e], d = dst[e];
  float w = rsqrtf(deg[s]) * rsqrtf(deg[d]);
#pragma unroll
  for (int i = 0; i < 4; ++i) {
    int c = lane + 32 * i;
    atomicAdd(&out[d * DD + c], xw[s * DD + c] * w);
  }
}

// relu + layernorm, one wave per 128-wide row; also emit f16 copy for WMMA consumers
__global__ void relu_ln_kernel(const float* __restrict__ in, const float* __restrict__ g,
                               const float* __restrict__ b, float* __restrict__ h,
                               _Float16* __restrict__ h16) {
  int wid = threadIdx.x >> 5, lane = threadIdx.x & 31;
  int row = blockIdx.x * (blockDim.x >> 5) + wid;
  if (row >= NN) return;
  float v[4], s1 = 0.f, s2 = 0.f;
#pragma unroll
  for (int i = 0; i < 4; ++i) {
    float x = fmaxf(in[row * DD + lane * 4 + i], 0.f);
    v[i] = x; s1 += x; s2 += x * x;
  }
  s1 = red32_sum(s1); s2 = red32_sum(s2);
  float mu = s1 * (1.f / DD);
  float var = s2 * (1.f / DD) - mu * mu;
  float inv = rsqrtf(var + 1e-5f);
#pragma unroll
  for (int i = 0; i < 4; ++i) {
    int c = lane * 4 + i;
    float y = (v[i] - mu) * inv * g[c] + b[c];
    h[row * DD + c] = y;
    h16[row * DD + c] = (_Float16)y;
  }
}

// ---------------- conversions ----------------

// WT[n][k] = (f16) W[k][n], W is [K][Nc]
__global__ void wt_convert_kernel(const float* __restrict__ W, _Float16* __restrict__ WT,
                                  int K, int Nc) {
  int idx = blockIdx.x * blockDim.x + threadIdx.x;
  if (idx >= K * Nc) return;
  int n = idx / K, k = idx % K;
  WT[idx] = (_Float16)W[k * Nc + n];
}

// posT[b][c][l] = (f16) pos_CB[b][l][c]
__global__ void post_convert_kernel(const float* __restrict__ pos, _Float16* __restrict__ posT) {
  int idx = blockIdx.x * blockDim.x + threadIdx.x;
  if (idx >= BN * 3 * LL) return;
  int b = idx / (3 * LL);
  int c = (idx / LL) % 3;
  int l = idx % LL;
  posT[idx] = (_Float16)pos[(b * LL + l) * 3 + c];
}

// ---------------- WMMA GEMM kernels ----------------

// q/k/v projections: per wave one 16x16 tile. grid.y = type*4 + head.
// q,k stored [B,H,L,16]; v stored transposed [B,H,16,L] for K-contiguous B-frags.
__global__ __launch_bounds__(32) void qkv_wmma_kernel(
    const _Float16* __restrict__ h16, const _Float16* __restrict__ WTq,
    const _Float16* __restrict__ WTk, const _Float16* __restrict__ WTv,
    _Float16* __restrict__ q16, _Float16* __restrict__ k16, _Float16* __restrict__ vT16) {
  int lane = threadIdx.x;
  int mt = blockIdx.x;
  int type = blockIdx.y >> 2, hh = blockIdx.y & 3;
  const _Float16* WT = (type == 0) ? WTq : (type == 1) ? WTk : WTv;
  const _Float16* BT = WT + (hh * 16) * DD;
  const _Float16* A = h16 + (mt * 16) * DD;
  v8f acc = {};
#pragma unroll
  for (int s = 0; s < 4; ++s) {
    v16h a = load_a_frag(A, DD, lane, s * 32, DD);
    v16h bb = load_b_frag(BT, DD, lane, s * 32, DD, 16);
    acc = wmma_f16(a, bb, acc);
  }
  int n = lane & 15, hi = lane >> 4;
  int row0 = mt * 16;
  int b = row0 >> 10, l0 = row0 & 1023;
#pragma unroll
  for (int r = 0; r < 8; ++r) {
    int l = l0 + r + 8 * hi;
    if (type == 0)      q16[((b * HH + hh) * LL + l) * DK + n] = (_Float16)acc[r];
    else if (type == 1) k16[((b * HH + hh) * LL + l) * DK + n] = (_Float16)acc[r];
    else                vT16[((b * HH + hh) * DK + n) * LL + l] = (_Float16)acc[r];
  }
}

// C[M,128] = A16[M,128] @ W (via WT[128][128]); used for GCN2 x@W
__global__ __launch_bounds__(32) void gemm128_wmma_kernel(const _Float16* __restrict__ A16,
                                                          const _Float16* __restrict__ WT,
                                                          float* __restrict__ C) {
  int lane = threadIdx.x;
  int mt = blockIdx.x, nt = blockIdx.y;
  const _Float16* A = A16 + (mt * 16) * DD;
  const _Float16* BT = WT + (nt * 16) * DD;
  v8f acc = {};
#pragma unroll
  for (int s = 0; s < 4; ++s) {
    v16h a = load_a_frag(A, DD, lane, s * 32, DD);
    v16h bb = load_b_frag(BT, DD, lane, s * 32, DD, 16);
    acc = wmma_f16(a, bb, acc);
  }
  int n = lane & 15, hi = lane >> 4;
#pragma unroll
  for (int r = 0; r < 8; ++r)
    C[(mt * 16 + r + 8 * hi) * DD + nt * 16 + n] = acc[r];
}

// ---------------- flash attention (one wave per (b,h,16-row tile)) ----------------
// Exploits: alpha @ rel_pos = alpha @ pos_CB - pos_CA  (alpha rows sum to 1).
// attnraw[b,l,h,0:16] = alpha@v ; attnraw[b,l,h,16:19] = alpha@pos_CB
__global__ __launch_bounds__(32) void flash_attn_kernel(
    const _Float16* __restrict__ q16, const _Float16* __restrict__ k16,
    const _Float16* __restrict__ vT16, const _Float16* __restrict__ posT16,
    float* __restrict__ attnraw) {
  __shared__ _Float16 Plds[16 * 32];
  int lane = threadIdx.x;
  int mt = blockIdx.x, hh = blockIdx.y, b = blockIdx.z;
  int head = b * HH + hh;
  const _Float16* qbase = q16 + (head * LL + mt * 16) * DK;
  const _Float16* kbase = k16 + head * LL * DK;
  const _Float16* vbase = vT16 + head * DK * LL;
  const _Float16* pbase = posT16 + b * 3 * LL;

  v16h aq = load_a_frag(qbase, DK, lane, 0, DK);  // 16 x (16, zero-padded to 32)
  v8f accV = {}, accP = {};
  float run_max[8], run_sum[8];
#pragma unroll
  for (int r = 0; r < 8; ++r) { run_max[r] = -1e30f; run_sum[r] = 0.f; }
  int hi = lane >> 4, nn = lane & 15;

  for (int kt = 0; kt < LL / 32; ++kt) {
    int kb = kt * 32;
    // scores for 32 keys (two 16-key sub-tiles); reduction dim dk=16 zero-padded
    v16h b0 = load_b_frag(kbase + kb * DK, DK, lane, 0, DK, 16);
    v16h b1 = load_b_frag(kbase + (kb + 16) * DK, DK, lane, 0, DK, 16);
    v8f z = {};
    v8f s0 = wmma_f16(aq, b0, z);
    v8f s1 = wmma_f16(aq, b1, z);
    // online softmax; rows live in 16-lane halves of the C layout
#pragma unroll
    for (int r = 0; r < 8; ++r) {
      float mx = red16_max(fmaxf(s0[r], s1[r]));
      float nm = fmaxf(run_max[r], mx);
      float sc = __expf(run_max[r] - nm);
      run_max[r] = nm;
      float p0 = __expf(s0[r] - nm);
      float p1 = __expf(s1[r] - nm);
      float ps = red16_sum(p0 + p1);
      run_sum[r] = run_sum[r] * sc + ps;
      accV[r] *= sc;
      accP[r] *= sc;
      int m = r + 8 * hi;
      Plds[m * 32 + nn] = (_Float16)p0;
      Plds[m * 32 + 16 + nn] = (_Float16)p1;
    }
    __syncthreads();  // single wave: ordering fence for LDS relayout
    v16h ap = load_a_frag(Plds, 32, lane, 0, 32);           // P: 16 x 32 keys
    v16h bv = load_b_frag(vbase, LL, lane, kb, LL, 16);     // V : 32 keys x 16
    v16h bp = load_b_frag(pbase, LL, lane, kb, LL, 3);      // pos: 32 keys x 3
    accV = wmma_f16(ap, bv, accV);
    accP = wmma_f16(ap, bp, accP);
    __syncthreads();
  }
#pragma unroll
  for (int r = 0; r < 8; ++r) {
    int m = r + 8 * hi;
    int l = mt * 16 + m;
    float inv = 1.f / run_sum[r];
    float* outp = attnraw + ((b * LL + l) * HH + hh) * 19;
    outp[nn] = accV[r] * inv;
    if (nn < 3) outp[16 + nn] = accP[r] * inv;
  }
}

// ---------------- spatial features + Wo + residual + LN ----------------
__global__ __launch_bounds__(128) void post_attn_kernel(
    const float* __restrict__ h_in, const float* __restrict__ attnraw,
    const float* __restrict__ posCA, const float* __restrict__ frame,
    const float* __restrict__ Wo, const float* __restrict__ bo,
    const float* __restrict__ lng, const float* __restrict__ lnb,
    float* __restrict__ h_out, _Float16* __restrict__ h_out16) {
  __shared__ float feat[92];
  __shared__ float red1[4], red2[4];
  int tok = blockIdx.x;
  int tid = threadIdx.x;
  const float* araw = attnraw + tok * (HH * 19);
  if (tid < 64) {
    feat[tid] = araw[(tid >> 4) * 19 + (tid & 15)];  // feat_node [h*16+d]
  } else if (tid < 68) {
    int hh2 = tid - 64;
    float bias[3];
#pragma unroll
    for (int c = 0; c < 3; ++c) bias[c] = araw[hh2 * 19 + 16 + c] - posCA[tok * 3 + c];
    float dist = sqrtf(bias[0] * bias[0] + bias[1] * bias[1] + bias[2] * bias[2]);
    float pts[3], pn2 = 0.f;
#pragma unroll
    for (int i = 0; i < 3; ++i) {
      float s = 0.f;
#pragma unroll
      for (int j = 0; j < 3; ++j) s += frame[tok * 9 + i * 3 + j] * bias[j];
      pts[i] = s; pn2 += s * s;
    }
    float pninv = 1.0f / (sqrtf(pn2) + 1e-10f);
#pragma unroll
    for (int i = 0; i < 3; ++i) {
      feat[64 + hh2 * 3 + i] = pts[i];          // pts block
      feat[80 + hh2 * 3 + i] = pts[i] * pninv;  // dirn block
    }
    feat[76 + hh2] = dist;                      // dist block
  }
  __syncthreads();
  float x = h_in[tok * DD + tid];
  float acc = bo[tid];
  for (int j = 0; j < 92; ++j) acc += feat[j] * Wo[j * DD + tid];
  float t = x + acc;
  // layernorm over 128 dims (4 waves)
  float s1 = red32_sum(t), s2 = red32_sum(t * t);
  int wid = tid >> 5, lane = tid & 31;
  if (lane == 0) { red1[wid] = s1; red2[wid] = s2; }
  __syncthreads();
  float S1 = red1[0] + red1[1] + red1[2] + red1[3];
  float S2 = red2[0] + red2[1] + red2[2] + red2[3];
  float mu = S1 * (1.f / DD);
  float var = S2 * (1.f / DD) - mu * mu;
  float y = (t - mu) * rsqrtf(var + 1e-5f) * lng[tid] + lnb[tid];
  float out = x + y;  // h = h + LN(h + feat)
  h_out[tok * DD + tid] = out;
  h_out16[tok * DD + tid] = (_Float16)out;
}

// ---------------- readout + MLP ----------------

__global__ void readout_kernel(const float* __restrict__ h, float* __restrict__ g) {
  int b = blockIdx.x, d = threadIdx.x;
  float s = 0.f;
  for (int l = 0; l < LL; ++l) s += h[(b * LL + l) * DD + d];
  g[b * DD + d] = s * (1.f / LL);
}

__global__ void fc1_kernel(const float* __restrict__ g, const float* __restrict__ W,
                           const float* __restrict__ bias, float* __restrict__ o) {
  int idx = blockIdx.x * blockDim.x + threadIdx.x;
  if (idx >= BN * 1024) return;
  int b = idx >> 10, j = idx & 1023;
  float acc = bias[j];
  for (int d = 0; d < DD; ++d) acc += g[b * DD + d] * W[d * 1024 + j];
  o[idx] = fmaxf(acc, 0.f);
}

__global__ void fc2_kernel(const float* __restrict__ o, const float* __restrict__ W,
                           const float* __restrict__ bias, float* __restrict__ out) {
  int idx = blockIdx.x * blockDim.x + threadIdx.x;
  if (idx >= BN * DD) return;
  int b = idx >> 7, d = idx & 127;
  float acc = bias[d];
  for (int j = 0; j < 1024; ++j) acc += o[b * 1024 + j] * W[j * DD + d];
  out[idx] = acc;
}

// ---------------- host ----------------

extern "C" void kernel_launch(void* const* d_in, const int* in_sizes, int n_in,
                              void* d_out, int out_size, void* d_ws, size_t ws_size,
                              hipStream_t stream) {
  (void)n_in; (void)out_size; (void)ws_size;
  const float* px    = (const float*)d_in[0];
  const int*   eidx  = (const int*)d_in[1];
  const float* posCA = (const float*)d_in[2];
  const float* posCB = (const float*)d_in[3];
  const float* frame = (const float*)d_in[4];
  const float* W_g1  = (const float*)d_in[5];
  const float* b_g1  = (const float*)d_in[6];
  const float* W_g2  = (const float*)d_in[7];
  const float* b_g2  = (const float*)d_in[8];
  const float* bn1_g = (const float*)d_in[9];
  const float* bn1_b = (const float*)d_in[10];
  const float* bn2_g = (const float*)d_in[11];
  const float* bn2_b = (const float*)d_in[12];
  const float* Wfc1  = (const float*)d_in[27];
  const float* bfc1  = (const float*)d_in[28];
  const float* Wfc2  = (const float*)d_in[29];
  const float* bfc2  = (const float*)d_in[30];

  int E = in_sizes[1] / 2;
  const int* src = eidx;
  const int* dst = eidx + E;

  char* ws = (char*)d_ws;
  size_t off = 0;
  auto alloc = [&](size_t bytes) -> void* {
    off = (off + 255) & ~size_t(255);
    void* p = ws + off;
    off += bytes;
    return p;
  };
  float*    xw      = (float*)alloc((size_t)NN * DD * 4);
  float*    agg     = (float*)alloc((size_t)NN * DD * 4);
  float*    h_a     = (float*)alloc((size_t)NN * DD * 4);
  float*    h_b     = (float*)alloc((size_t)NN * DD * 4);
  _Float16* h16     = (_Float16*)alloc((size_t)NN * DD * 2);
  _Float16* hb16    = (_Float16*)alloc((size_t)NN * DD * 2);
  float*    deg     = (float*)alloc((size_t)NN * 4);
  _Float16* q16     = (_Float16*)alloc((size_t)NN * 64 * 2);
  _Float16* k16     = (_Float16*)alloc((size_t)NN * 64 * 2);
  _Float16* vT16    = (_Float16*)alloc((size_t)NN * 64 * 2);
  _Float16* posT    = (_Float16*)alloc((size_t)BN * 3 * LL * 2);
  float*    attnraw = (float*)alloc((size_t)NN * HH * 19 * 4);
  _Float16* WTq     = (_Float16*)alloc((size_t)64 * DD * 2);
  _Float16* WTk     = (_Float16*)alloc((size_t)64 * DD * 2);
  _Float16* WTv     = (_Float16*)alloc((size_t)64 * DD * 2);
  _Float16* WgT     = (_Float16*)alloc((size_t)DD * DD * 2);
  float*    gvec    = (float*)alloc((size_t)BN * DD * 4);
  float*    fc1o    = (float*)alloc((size_t)BN * 1024 * 4);

  dim3 b256(256);

  // degrees (shared by both GCN layers) and pos transpose (shared by both attn layers)
  deg_init_kernel<<<(NN + 255) / 256, b256, 0, stream>>>(deg);
  deg_accum_kernel<<<(E + 255) / 256, b256, 0, stream>>>(dst, deg, E);
  post_convert_kernel<<<(BN * 3 * LL + 255) / 256, b256, 0, stream>>>(posCB, posT);

  // GCN layer 1
  gcn1_xw_kernel<<<(NN * DD + 255) / 256, b256, 0, stream>>>(px, W_g1, xw, NN * DD);
  gcn_selfbias_kernel<<<(NN * DD + 255) / 256, b256, 0, stream>>>(xw, deg, b_g1, agg, NN * DD);
  gcn_scatter_kernel<<<((unsigned)E * 32 + 255) / 256, b256, 0, stream>>>(src, dst, xw, deg, agg, E);
  relu_ln_kernel<<<NN / 8, b256, 0, stream>>>(agg, bn1_g, bn1_b, h_a, h16);

  auto run_attn = [&](const float* hin, const _Float16* hin16, int pbase,
                      float* hout, _Float16* hout16) {
    const float* Wq = (const float*)d_in[pbase + 0];
    const float* Wk = (const float*)d_in[pbase + 1];
    const float* Wv = (const float*)d_in[pbase + 2];
    const float* Wo = (const float*)d_in[pbase + 3];
    const float* bo = (const float*)d_in[pbase + 4];
    const float* lg = (const float*)d_in[pbase + 5];
    const float* lb = (const float*)d_in[pbase + 6];
    wt_convert_kernel<<<(DD * 64 + 255) / 256, b256, 0, stream>>>(Wq, WTq, DD, 64);
    wt_convert_kernel<<<(DD * 64 + 255) / 256, b256, 0, stream>>>(Wk, WTk, DD, 64);
    wt_convert_kernel<<<(DD * 64 + 255) / 256, b256, 0, stream>>>(Wv, WTv, DD, 64);
    qkv_wmma_kernel<<<dim3(NN / 16, 12), dim3(32), 0, stream>>>(hin16, WTq, WTk, WTv,
                                                                q16, k16, vT16);
    flash_attn_kernel<<<dim3(LL / 16, HH, BN), dim3(32), 0, stream>>>(q16, k16, vT16, posT,
                                                                      attnraw);
    post_attn_kernel<<<NN, dim3(128), 0, stream>>>(hin, attnraw, posCA, frame, Wo, bo, lg, lb,
                                                   hout, hout16);
  };

  run_attn(h_a, h16, 13, h_b, hb16);

  // GCN layer 2 (WMMA GEMM for x@W_g2)
  wt_convert_kernel<<<(DD * DD + 255) / 256, b256, 0, stream>>>(W_g2, WgT, DD, DD);
  gemm128_wmma_kernel<<<dim3(NN / 16, DD / 16), dim3(32), 0, stream>>>(hb16, WgT, xw);
  gcn_selfbias_kernel<<<(NN * DD + 255) / 256, b256, 0, stream>>>(xw, deg, b_g2, agg, NN * DD);
  gcn_scatter_kernel<<<((unsigned)E * 32 + 255) / 256, b256, 0, stream>>>(src, dst, xw, deg, agg, E);
  relu_ln_kernel<<<NN / 8, b256, 0, stream>>>(agg, bn2_g, bn2_b, h_a, h16);

  run_attn(h_a, h16, 20, h_b, hb16);

  // readout + MLP
  readout_kernel<<<BN, dim3(DD), 0, stream>>>(h_b, gvec);
  fc1_kernel<<<(BN * 1024 + 255) / 256, b256, 0, stream>>>(gvec, Wfc1, bfc1, fc1o);
  fc2_kernel<<<(BN * DD + 255) / 256, b256, 0, stream>>>(fc1o, Wfc2, bfc2, (float*)d_out);
}
